// MambaAuroraPredictor_44409961840738
// MI455X (gfx1250) — matrix operand order repaired and checked by reading, not compiled, not run on hardware
//
#include <hip/hip_runtime.h>
#include <hip/hip_bf16.h>

typedef __attribute__((ext_vector_type(16))) _Float16 v16h;
typedef __attribute__((ext_vector_type(8)))  _Float16 v8h;
typedef __attribute__((ext_vector_type(8)))  float    v8f;

#define NPIX 16384   // 128*128
#define HID  128
#define DI   256
#define DS   16
#define DTR  8
#define PRLD 64      // padded xproj output width

// ---------------------------------------------------------------- helpers
__device__ __forceinline__ float gelu_f(float v) {
    return 0.5f * v * (1.0f + erff(v * 0.70710678118f));
}
__device__ __forceinline__ float silu_f(float v) {
    return v / (1.0f + __expf(-v));
}

// two contiguous 8-half runs -> one v16h fragment (2x global_load_b128)
__device__ __forceinline__ v16h frag16(const _Float16* __restrict__ p0,
                                       const _Float16* __restrict__ p1) {
    v8h lo = *(const v8h*)p0;
    v8h hi = *(const v8h*)p1;
    return __builtin_shufflevector(lo, hi, 0, 1, 2, 3, 4, 5, 6, 7,
                                   8, 9, 10, 11, 12, 13, 14, 15);
}

// A fragment: 16x32 f16. lane m = lane&15, hi = lane>>4;
// halfs = A[m][k0 + hi*8 + 0..7] ++ A[m][k0 + 16 + hi*8 + 0..7]
__device__ __forceinline__ v16h load_a(const _Float16* __restrict__ rowk0) {
    int h8 = ((threadIdx.x >> 4) & 1) * 8;
    return frag16(rowk0 + h8, rowk0 + 16 + h8);
}
__device__ __forceinline__ v16h load_a_guard(const _Float16* __restrict__ rowk0, bool valid) {
    if (valid) return load_a(rowk0);
    v16h z = {};
    return z;
}

// B fragment: 32x16 f16 from f16 weight W[N][K] (A @ W^T), unconditional.
// lane n = n0 + lane&15; halfs = W[n][k0 + hi*8 + 0..7] ++ W[n][k0 + 16 + hi*8 + 0..7]
__device__ __forceinline__ v16h load_b(const _Float16* __restrict__ W, int ldk,
                                       int n0, int k0) {
    const _Float16* p = W + (size_t)(n0 + (threadIdx.x & 15)) * ldk + k0 +
                        ((threadIdx.x >> 4) & 1) * 8;
    return frag16(p, p + 16);
}

__device__ __forceinline__ v8f wmma_f16(v16h a, v16h b, v8f c) {
    return __builtin_amdgcn_wmma_f32_16x16x32_f16(false, a, false, b,
                                                  (short)0, c, false, false);
}

// ---------------------------------------------------------------- weight prep
__global__ __launch_bounds__(256)
void cvt_w_kernel(const float* __restrict__ src, _Float16* __restrict__ dst, int n) {
    int i = blockIdx.x * blockDim.x + threadIdx.x;
    if (i < n) dst[i] = (_Float16)src[i];
}

// xproj [3][40][256] f32 -> padded [3][64][256] f16 (rows 40..63 = 0)
__global__ __launch_bounds__(256)
void xproj_pad_kernel(const float* __restrict__ src, _Float16* __restrict__ dst) {
    int i = blockIdx.x * blockDim.x + threadIdx.x;   // 3*64*256
    int k = i & 255;
    int n = (i >> 8) & 63;
    int l = i >> 14;
    dst[i] = (n < 40) ? (_Float16)src[(size_t)l * 40 * 256 + n * 256 + k]
                      : (_Float16)0.0f;
}

// conv weight [oc][ic][3x3] f32 -> [tap][oc][ic] f16 (B fragment contiguous in k=ic)
__global__ __launch_bounds__(256)
void conv_w_repack_kernel(const float* __restrict__ w, _Float16* __restrict__ out) {
    int i = blockIdx.x * blockDim.x + threadIdx.x;   // 9*128*128
    int tap = i >> 14;
    int oc  = (i >> 7) & 127;
    int ic  = i & 127;
    out[i] = (_Float16)w[oc * 1152 + ic * 9 + tap];
}

// ---------------------------------------------------------------- enc1 (1->128) + GELU -> f16
__global__ __launch_bounds__(256)
void enc1_gelu_kernel(const float* __restrict__ x, const float* __restrict__ w,
                      const float* __restrict__ b, _Float16* __restrict__ out) {
    int idx = blockIdx.x * blockDim.x + threadIdx.x;      // NPIX*128
    int p = idx >> 7, oc = idx & 127;
    int yy = p >> 7, xx = p & 127;
    float acc = b[oc];
#pragma unroll
    for (int dy = -1; dy <= 1; ++dy) {
        int sy = yy + dy;
        if (sy < 0 || sy >= 128) continue;
#pragma unroll
        for (int dx = -1; dx <= 1; ++dx) {
            int sx = xx + dx;
            if (sx < 0 || sx >= 128) continue;
            acc += x[sy * 128 + sx] * w[oc * 9 + (dy + 1) * 3 + (dx + 1)];
        }
    }
    out[(size_t)p * HID + oc] = (_Float16)gelu_f(acc);
}

// ---------------------------------------------------------------- 128->128 3x3 conv, 2x2 register tile
// in: f16 [NPIX][128]; wp: packed f16 [9][128][128]; out: f16 [NPIX][128]
template <bool GELU>
__global__ __launch_bounds__(256)
void conv3x3_gemm_kernel(const _Float16* __restrict__ in, const _Float16* __restrict__ wp,
                         const float* __restrict__ bias, _Float16* __restrict__ out) {
    int wave = blockIdx.x * (blockDim.x >> 5) + (threadIdx.x >> 5);
    int mt = wave >> 2;            // 4 n-supertiles (128/32)
    int nt = wave & 3;
    int m0 = mt * 32, n0 = nt * 32;
    int lane15 = threadIdx.x & 15;
    int ma = m0 + lane15;          // sub-tile 0 pixel
    int mb = ma + 16;              // sub-tile 1 pixel (same image row: 32 | 128)
    int yy = ma >> 7;
    int xa = ma & 127, xb = mb & 127;

    v8f a00 = {}, a01 = {}, a10 = {}, a11 = {};
#pragma unroll
    for (int dy = -1; dy <= 1; ++dy) {
        int sy = yy + dy;
        bool yok = (sy >= 0) && (sy < 128);
#pragma unroll
        for (int dx = -1; dx <= 1; ++dx) {
            bool v0 = yok && (xa + dx >= 0) && (xa + dx < 128);
            bool v1 = yok && (xb + dx >= 0) && (xb + dx < 128);
            const _Float16* r0 = in + (size_t)(sy * 128 + xa + dx) * HID;
            const _Float16* r1 = in + (size_t)(sy * 128 + xb + dx) * HID;
            const _Float16* wtap = wp + (size_t)((dy + 1) * 3 + (dx + 1)) * (128 * 128);
#pragma unroll
            for (int k0 = 0; k0 < HID; k0 += 32) {
                v16h fa0 = load_a_guard(r0 + k0, v0);
                v16h fa1 = load_a_guard(r1 + k0, v1);
                v16h fb0 = load_b(wtap, HID, n0, k0);
                v16h fb1 = load_b(wtap, HID, n0 + 16, k0);
                a00 = wmma_f16(fa0, fb0, a00);
                a01 = wmma_f16(fa0, fb1, a01);
                a10 = wmma_f16(fa1, fb0, a10);
                a11 = wmma_f16(fa1, fb1, a11);
            }
        }
    }
    int n  = n0 + lane15;
    int hi = (threadIdx.x >> 4) & 1;
    float bn0 = bias[n], bn1 = bias[n + 16];
#pragma unroll
    for (int r = 0; r < 8; ++r) {
        int mr0 = m0 + hi * 8 + r, mr1 = mr0 + 16;
        float v;
        v = a00[r] + bn0; if (GELU) v = gelu_f(v);
        out[(size_t)mr0 * HID + n] = (_Float16)v;
        v = a01[r] + bn1; if (GELU) v = gelu_f(v);
        out[(size_t)mr0 * HID + n + 16] = (_Float16)v;
        v = a10[r] + bn0; if (GELU) v = gelu_f(v);
        out[(size_t)mr1 * HID + n] = (_Float16)v;
        v = a11[r] + bn1; if (GELU) v = gelu_f(v);
        out[(size_t)mr1 * HID + n + 16] = (_Float16)v;
    }
}

// ---------------------------------------------------------------- generic C[M,N] = A[M,K] @ W[N,K]^T, 2x2 tile
template <typename OutT, int K>
__global__ __launch_bounds__(256)
void gemm_bT_kernel(const _Float16* __restrict__ A, const _Float16* __restrict__ W,
                    OutT* __restrict__ C, int ldc, int ntiles2) {
    int wave = blockIdx.x * (blockDim.x >> 5) + (threadIdx.x >> 5);
    int mt = wave / ntiles2;
    int nt = wave - mt * ntiles2;
    int m0 = mt * 32, n0 = nt * 32;
    int lane15 = threadIdx.x & 15;

    const _Float16* ar0 = A + (size_t)(m0 + lane15) * K;
    const _Float16* ar1 = ar0 + (size_t)16 * K;
    v8f a00 = {}, a01 = {}, a10 = {}, a11 = {};
#pragma unroll
    for (int k0 = 0; k0 < K; k0 += 32) {
        v16h fa0 = load_a(ar0 + k0);
        v16h fa1 = load_a(ar1 + k0);
        v16h fb0 = load_b(W, K, n0, k0);
        v16h fb1 = load_b(W, K, n0 + 16, k0);
        a00 = wmma_f16(fa0, fb0, a00);
        a01 = wmma_f16(fa0, fb1, a01);
        a10 = wmma_f16(fa1, fb0, a10);
        a11 = wmma_f16(fa1, fb1, a11);
    }
    int n  = n0 + lane15;
    int hi = (threadIdx.x >> 4) & 1;
#pragma unroll
    for (int r = 0; r < 8; ++r) {
        int mr0 = m0 + hi * 8 + r, mr1 = mr0 + 16;
        C[(size_t)mr0 * ldc + n]      = (OutT)a00[r];
        C[(size_t)mr0 * ldc + n + 16] = (OutT)a01[r];
        C[(size_t)mr1 * ldc + n]      = (OutT)a10[r];
        C[(size_t)mr1 * ldc + n + 16] = (OutT)a11[r];
    }
}

// ---------------------------------------------------------------- depthwise causal conv1d + bias + SiLU -> f16
__global__ __launch_bounds__(256)
void dwconv_silu_kernel(const float* __restrict__ XZ, const float* __restrict__ cw,
                        const float* __restrict__ cb, _Float16* __restrict__ U) {
    int idx = blockIdx.x * blockDim.x + threadIdx.x;   // NPIX * DI
    int row = idx >> 8, c = idx & 255;
    int t = row & 127;
    float acc = cb[c];
#pragma unroll
    for (int j = 0; j < 4; ++j) {
        int tt = t + j - 3;
        if (tt >= 0) acc += cw[c * 4 + j] * XZ[(size_t)(row + j - 3) * 512 + c];
    }
    U[(size_t)row * DI + c] = (_Float16)silu_f(acc);
}

// ---------------------------------------------------------------- fused selective scan
__global__ __launch_bounds__(256)
void scan_kernel(const float* __restrict__ XZ, const _Float16* __restrict__ U,
                 const float* __restrict__ PR, const float* __restrict__ dt_w,
                 const float* __restrict__ dt_b, const float* __restrict__ Alog,
                 const float* __restrict__ Dp, _Float16* __restrict__ Y) {
    int s = blockIdx.x;
    int c = threadIdx.x;
    __shared__ float sm[40];                 // dt[8] | B[16] | C[16]

    float dtw[DTR];
#pragma unroll
    for (int j = 0; j < DTR; ++j) dtw[j] = dt_w[c * DTR + j];
    float dtb = dt_b[c];
    float Dv  = Dp[c];
    float An[DS];
#pragma unroll
    for (int j = 0; j < DS; ++j) An[j] = -__expf(Alog[c * DS + j]);
    float h[DS];
#pragma unroll
    for (int j = 0; j < DS; ++j) h[j] = 0.0f;

    for (int t = 0; t < 128; ++t) {
        int row = (s << 7) + t;
        __syncthreads();
        if (threadIdx.x < 40) sm[threadIdx.x] = PR[(size_t)row * PRLD + threadIdx.x];
        __syncthreads();

        float acc = dtb;
#pragma unroll
        for (int j = 0; j < DTR; ++j) acc += sm[j] * dtw[j];
        float delta = (acc > 20.0f) ? acc : log1pf(__expf(acc));

        float uv = (float)U[(size_t)row * DI + c];
        float du = delta * uv;
        float yv = 0.0f;
#pragma unroll
        for (int j = 0; j < DS; ++j) {
            float dA = __expf(delta * An[j]);
            h[j] = dA * h[j] + du * sm[8 + j];
            yv += h[j] * sm[24 + j];
        }
        float zv = XZ[(size_t)row * 512 + DI + c];
        Y[(size_t)row * DI + c] = (_Float16)((yv + uv * Dv) * silu_f(zv));
    }
}

// ---------------------------------------------------------------- layout shuffles (f16)
__global__ __launch_bounds__(256)
void transpose_hw_kernel(const _Float16* __restrict__ in, _Float16* __restrict__ out) {
    int idx = blockIdx.x * blockDim.x + threadIdx.x;   // NPIX*HID
    int p = idx >> 7, c = idx & 127;
    int yy = p >> 7, xx = p & 127;
    out[(size_t)((xx << 7) | yy) * HID + c] = in[(size_t)p * HID + c];
}

__global__ __launch_bounds__(256)
void combine_kernel(const _Float16* __restrict__ outH, const _Float16* __restrict__ outW,
                    _Float16* __restrict__ dst) {
    int idx = blockIdx.x * blockDim.x + threadIdx.x;
    int p = idx >> 7, c = idx & 127;
    int yy = p >> 7, xx = p & 127;
    dst[(size_t)p * HID + c] =
        (_Float16)((float)outH[(size_t)((xx << 7) | yy) * HID + c] +
                   (float)outW[(size_t)p * HID + c]);
}

// ---------------------------------------------------------------- pred2 (128->1)
__global__ __launch_bounds__(256)
void pred2_kernel(const _Float16* __restrict__ in, const float* __restrict__ w,
                  const float* __restrict__ b, float* __restrict__ out) {
    int p = blockIdx.x * blockDim.x + threadIdx.x;   // NPIX
    int yy = p >> 7, xx = p & 127;
    float acc = b[0];
    for (int dy = -1; dy <= 1; ++dy) {
        int sy = yy + dy;
        if (sy < 0 || sy >= 128) continue;
        for (int dx = -1; dx <= 1; ++dx) {
            int sx = xx + dx;
            if (sx < 0 || sx >= 128) continue;
            const _Float16* row = in + (size_t)(sy * 128 + sx) * HID;
            int tap = (dy + 1) * 3 + (dx + 1);
#pragma unroll
            for (int ic = 0; ic < HID; ++ic) acc += (float)row[ic] * w[ic * 9 + tap];
        }
    }
    out[p] = acc;
}

// ---------------------------------------------------------------- host
extern "C" void kernel_launch(void* const* d_in, const int* in_sizes, int n_in,
                              void* d_out, int out_size, void* d_ws, size_t ws_size,
                              hipStream_t stream) {
    const float* x       = (const float*)d_in[0];
    const float* enc1_w  = (const float*)d_in[1];
    const float* enc1_b  = (const float*)d_in[2];
    const float* enc2_w  = (const float*)d_in[3];
    const float* enc2_b  = (const float*)d_in[4];
    const float* m_in_w  = (const float*)d_in[5];
    const float* m_cw    = (const float*)d_in[6];
    const float* m_cb    = (const float*)d_in[7];
    const float* m_xpw   = (const float*)d_in[8];
    const float* m_dtw   = (const float*)d_in[9];
    const float* m_dtb   = (const float*)d_in[10];
    const float* m_Alog  = (const float*)d_in[11];
    const float* m_D     = (const float*)d_in[12];
    const float* m_outw  = (const float*)d_in[13];
    const float* fus_w   = (const float*)d_in[14];
    const float* fus_b   = (const float*)d_in[15];
    const float* pred1_w = (const float*)d_in[16];
    const float* pred1_b = (const float*)d_in[17];
    const float* pred2_w = (const float*)d_in[18];
    const float* pred2_b = (const float*)d_in[19];
    (void)ws_size; (void)in_sizes; (void)n_in; (void)out_size;

    size_t off = 0;
    auto alloc = [&](size_t bytes) {
        void* p = (char*)d_ws + off;
        off += (bytes + 255) & ~(size_t)255;
        return p;
    };
    // f16 activation buffers
    _Float16* T1 = (_Float16*)alloc((size_t)NPIX * HID * 2);
    _Float16* FT = (_Float16*)alloc((size_t)NPIX * HID * 2);
    _Float16* S0 = (_Float16*)alloc((size_t)NPIX * HID * 2);
    _Float16* S1 = (_Float16*)alloc((size_t)NPIX * HID * 2);
    _Float16* S2 = (_Float16*)alloc((size_t)NPIX * HID * 2);
    _Float16* U  = (_Float16*)alloc((size_t)NPIX * DI * 2);
    _Float16* Y  = (_Float16*)alloc((size_t)NPIX * DI * 2);
    // f32 scan-side buffers
    float* XZ = (float*)alloc((size_t)NPIX * 512 * 4);
    float* PR = (float*)alloc((size_t)NPIX * PRLD * 4);
    // f16 weights
    _Float16* INW  = (_Float16*)alloc((size_t)3 * 512 * 128 * 2);
    _Float16* XPW  = (_Float16*)alloc((size_t)3 * 64 * 256 * 2);   // padded N=64
    _Float16* OUTW = (_Float16*)alloc((size_t)3 * 128 * 256 * 2);
    _Float16* W2P  = (_Float16*)alloc((size_t)9 * 128 * 128 * 2);
    _Float16* WFP  = (_Float16*)alloc((size_t)9 * 128 * 128 * 2);
    _Float16* WP1  = (_Float16*)alloc((size_t)9 * 128 * 128 * 2);

    // ---- weight prep
    cvt_w_kernel<<<(3 * 512 * 128 + 255) / 256, 256, 0, stream>>>(m_in_w, INW, 3 * 512 * 128);
    xproj_pad_kernel<<<(3 * 64 * 256 + 255) / 256, 256, 0, stream>>>(m_xpw, XPW);
    cvt_w_kernel<<<(3 * 128 * 256 + 255) / 256, 256, 0, stream>>>(m_outw, OUTW, 3 * 128 * 256);
    conv_w_repack_kernel<<<576, 256, 0, stream>>>(enc2_w, W2P);
    conv_w_repack_kernel<<<576, 256, 0, stream>>>(fus_w, WFP);
    conv_w_repack_kernel<<<576, 256, 0, stream>>>(pred1_w, WP1);

    auto mamba = [&](const _Float16* Xin, _Float16* Xout, int l) {
        const _Float16* inw = INW + (size_t)l * 512 * 128;
        const _Float16* xpw = XPW + (size_t)l * 64 * 256;
        const _Float16* ow  = OUTW + (size_t)l * 128 * 256;
        const float* cw   = m_cw   + (size_t)l * DI * 4;
        const float* cb   = m_cb   + (size_t)l * DI;
        const float* dtw  = m_dtw  + (size_t)l * DI * DTR;
        const float* dtb  = m_dtb  + (size_t)l * DI;
        const float* alog = m_Alog + (size_t)l * DI * DS;
        const float* dp   = m_D    + (size_t)l * DI;
        // XZ = X @ in_w^T : 512x16 supertiles -> 1024 blocks
        gemm_bT_kernel<float, 128><<<1024, 256, 0, stream>>>(Xin, inw, XZ, 512, 16);
        dwconv_silu_kernel<<<NPIX, 256, 0, stream>>>(XZ, cw, cb, U);
        // proj = u @ xproj^T : N padded to 64 -> 512x2 supertiles -> 128 blocks
        gemm_bT_kernel<float, 256><<<128, 256, 0, stream>>>(U, xpw, PR, PRLD, 2);
        scan_kernel<<<128, 256, 0, stream>>>(XZ, U, PR, dtw, dtb, alog, dp, Y);
        // out = y @ out_w^T : 512x4 supertiles -> 256 blocks
        gemm_bT_kernel<_Float16, 256><<<256, 256, 0, stream>>>(Y, ow, Xout, 128, 4);
    };

    // encoder
    enc1_gelu_kernel<<<8192, 256, 0, stream>>>(x, enc1_w, enc1_b, T1);
    conv3x3_gemm_kernel<false><<<256, 256, 0, stream>>>(T1, W2P, enc2_b, FT);

    // h-direction scan (sequences = columns)
    transpose_hw_kernel<<<8192, 256, 0, stream>>>(FT, S0);
    mamba(S0, S1, 0);
    mamba(S1, S0, 1);
    mamba(S0, S1, 2);          // outH = S1 (seq layout)

    // w-direction scan (sequences = rows; feats already row-major)
    mamba(FT, S0, 0);
    mamba(S0, S2, 1);
    mamba(S2, S0, 2);          // outW = S0 (pixel layout)

    combine_kernel<<<8192, 256, 0, stream>>>(S1, S0, T1);

    // fuse + prediction head
    conv3x3_gemm_kernel<false><<<256, 256, 0, stream>>>(T1, WFP, fus_b, S2);
    conv3x3_gemm_kernel<true><<<256, 256, 0, stream>>>(S2, WP1, pred1_b, S0);
    pred2_kernel<<<64, 256, 0, stream>>>(S0, pred2_w, pred2_b, (float*)d_out);
}
